// SAM_79182017069396
// MI455X (gfx1250) — matrix operand order repaired and checked by reading, not compiled, not run on hardware
//
#include <hip/hip_runtime.h>
#include <hip/hip_bf16.h>

// ---------------------------------------------------------------------------
// Shapes: x (8, 256, 64, 64) fp32 -> B=8, C=256, S=4096, L=32 (latent)
// Pass 1: projections -> bf16 q/k/v in WMMA-friendly layouts (workspace);
//         q is pre-scaled by 1/sqrt(32).
// Pass 2: flash attention, 64 t-columns per iteration: 8 waves/block
//         (wave pair = one 16-row s-tile, each wave owns 128 of 256 output
//         channels). K/V chunks double-buffered in LDS via
//         global_load_async_to_lds_b128 (ASYNCcnt); logits, row-sums and
//         P*V on v_wmma_f32_16x16x32_bf16 (22 WMMAs / wave / iteration).
// ---------------------------------------------------------------------------

typedef __attribute__((ext_vector_type(16))) __bf16 v16bf;
typedef __attribute__((ext_vector_type(8)))  __bf16 v8bf;
typedef __attribute__((ext_vector_type(8)))  float  v8f;

#define SAM_S 4096
#define SAM_C 256
#define SAM_L 32

// ---------------------------------------------------------------------------
// Pass 1: q[l,s] (pre-scaled), k[l,t], v[c,t] projections (VALU; tiny FLOPs).
//   qs[b][s][l]  (bf16, s-major)  -- A-fragment source for Q
//   kt[b][t][l]  (bf16, t-major)  -- B-fragment source for K
//   vc[b][c][t]  (bf16, c-major)  -- B-fragment source for V
// ---------------------------------------------------------------------------
__global__ __launch_bounds__(256) void SAM_proj_kernel(
    const float* __restrict__ x, const float* __restrict__ w_qk,
    const float* __restrict__ w_v,
    unsigned short* __restrict__ qs_, unsigned short* __restrict__ kt_,
    unsigned short* __restrict__ vc_)
{
    __shared__ float xt[SAM_C][64];   // 64 KB
    __bf16* qs = (__bf16*)qs_;
    __bf16* kt = (__bf16*)kt_;
    __bf16* vc = (__bf16*)vc_;

    const int tid = threadIdx.x;
    const int b   = blockIdx.y;
    const int s0  = blockIdx.x * 64;
    const float* xb = x + (size_t)b * SAM_C * SAM_S;

    #pragma unroll 4
    for (int i = 0; i < 64; ++i) {
        int f  = tid + i * 256;
        int c  = f >> 6;
        int sl = f & 63;
        xt[c][sl] = xb[(size_t)c * SAM_S + s0 + sl];
    }
    __syncthreads();

    const int sl = tid & 63;
    const int og = tid >> 6;         // wave-uniform output group
    const int s  = s0 + sl;
    const float logit_scale = 0.17677669529663687f;   // 1/sqrt(32)

    for (int i = 0; i < 80; ++i) {
        int o = og * 80 + i;         // 0..319
        const float* wr = (o < 64) ? (w_qk + (size_t)o * SAM_C)
                                   : (w_v  + (size_t)(o - 64) * SAM_C);
        float acc = 0.0f;
        #pragma unroll 8
        for (int c = 0; c < SAM_C; ++c)
            acc = fmaf(wr[c], xt[c][sl], acc);

        if (o < SAM_L) {
            qs[((size_t)b * SAM_S + s) * SAM_L + o] = (__bf16)(acc * logit_scale);
        } else if (o < 2 * SAM_L) {
            kt[((size_t)b * SAM_S + s) * SAM_L + (o - SAM_L)] = (__bf16)acc;
        } else {
            vc[((size_t)b * SAM_C + (o - 64)) * (size_t)SAM_S + s] = (__bf16)acc;
        }
    }
}

// ---------------------------------------------------------------------------
static __device__ inline void async_ld16(unsigned lds_off, unsigned goff,
                                         unsigned long long base) {
    // GVS mode: mem = SADDR + VADDR32 ; LDS dest offset in first VGPR.
    asm volatile("global_load_async_to_lds_b128 %0, %1, %2"
                 :: "v"(lds_off), "v"(goff), "s"(base) : "memory");
}

// ---------------------------------------------------------------------------
// Pass 2: flash attention, 64 t per iteration. Block = 256 threads (8 waves):
//   s-tile = blockIdx.x*64 + (w>>1)*16 ; channel half = (w&1)*128.
// Fragment layouts (wave32, CDNA5 ISA 7.12.2):
//   A 16x32 bf16 : lane(lo,hi) row M=lo, elems K = 16hi+0..15 (contiguous order used here)
//   B 32x16 bf16 : lane col N=lo, elems K = 16hi+0..15 (contiguous order chosen here to match A)
//   C/D 16x16 f32: lane col N=lo, vgpr r -> row M = r + 8hi
// ---------------------------------------------------------------------------
__global__ __launch_bounds__(256) void SAM_attn_kernel(
    const float* __restrict__ x, const float* __restrict__ gamma,
    const unsigned short* __restrict__ qs_, const unsigned short* __restrict__ kt_,
    const unsigned short* __restrict__ vc_, float* __restrict__ out)
{
    __shared__ __align__(32) __bf16 ktile[2][64][32];     //  8 KB (dbl-buffered)
    __shared__ __align__(32) __bf16 vtile[2][SAM_C][64];  // 64 KB (dbl-buffered)
    __shared__ __align__(32) __bf16 pbuf[8][16][64];      // 16 KB (per-wave P)

    const __bf16* qs = (const __bf16*)qs_;
    const __bf16* kt = (const __bf16*)kt_;
    const __bf16* vc = (const __bf16*)vc_;

    const int tid  = threadIdx.x;
    const int lane = tid & 31;
    const int w    = tid >> 5;           // wave id 0..7
    const int lo   = lane & 15;
    const int hi   = lane >> 4;
    const int b    = blockIdx.y;
    const int s0   = blockIdx.x * 64 + (w >> 1) * 16;
    const int c0   = (w & 1) * 128;      // channel half owned by this wave

    const __bf16* qb = qs + (size_t)b * SAM_S * SAM_L;
    const unsigned long long kbb =
        (unsigned long long)(uintptr_t)(kt + (size_t)b * SAM_S * SAM_L);
    const unsigned long long vbb =
        (unsigned long long)(uintptr_t)(vc + (size_t)b * SAM_C * SAM_S);

    // ---- Q fragment (A-layout), loaded once (q pre-scaled by 1/sqrt(32)) ----
    v16bf qa;
    {
        const __bf16* p = qb + (size_t)(s0 + lo) * SAM_L + hi * 16;
        v8bf q0 = *(const v8bf*)(p);
        v8bf q1 = *(const v8bf*)(p + 8);
        #pragma unroll
        for (int i = 0; i < 8; ++i) { qa[i] = q0[i]; qa[8 + i] = q1[i]; }
    }

    v16bf ones;
    #pragma unroll
    for (int i = 0; i < 16; ++i) ones[i] = (__bf16)1.0f;

    v8f zero = {};
    v8f acc[8];
    #pragma unroll
    for (int j = 0; j < 8; ++j) acc[j] = zero;

    float mrow[8], lrow[8];
    #pragma unroll
    for (int r = 0; r < 8; ++r) { mrow[r] = -1e30f; lrow[r] = 0.0f; }

    // ---- async staging of one 64-wide t-chunk of K and V into LDS ----
    // V: 256 rows x 128 B (2048 x b128) ; K: 64 rows x 64 B (256 x b128).
    auto issue_chunk = [&](int t0c, int buf) {
        const unsigned vbase = (unsigned)(uintptr_t)&vtile[buf][0][0];
        const unsigned kbase = (unsigned)(uintptr_t)&ktile[buf][0][0];
        #pragma unroll
        for (int n = 0; n < 9; ++n) {
            int idx = tid + n * 256;             // 0..2303
            if (idx < 2048) {
                int c = idx >> 3, part = (idx & 7) * 16;
                async_ld16(vbase + c * 128 + part,
                           (unsigned)((c * SAM_S + t0c) * 2) + part, vbb);
            } else {
                int r = (idx - 2048) >> 2, part = (idx & 3) * 16;
                async_ld16(kbase + r * 64 + part,
                           (unsigned)((t0c + r) * SAM_L * 2) + part, kbb);
            }
        }
    };

    issue_chunk(0, 0);
    int cur = 0;

    for (int t0 = 0; t0 < SAM_S; t0 += 64) {
        asm volatile("s_wait_asynccnt 0x0" ::: "memory");
        __syncthreads();                       // tiles landed; prev buf free
        if (t0 + 64 < SAM_S) issue_chunk(t0 + 64, cur ^ 1);

        // ---- logits: four 16x16 tiles (K-dim = L = 32) from LDS K tile ----
        v8f sf[4];
        #pragma unroll
        for (int q4 = 0; q4 < 4; ++q4) {
            v16bf kf = *(const v16bf*)&ktile[cur][16 * q4 + lo][16 * hi];
            sf[q4] = __builtin_amdgcn_wmma_f32_16x16x32_bf16(
                false, qa, false, kf, (short)0, zero, false, false);
        }

        // ---- online softmax: row max over 64 cols ----
        float mloc[8];
        #pragma unroll
        for (int r = 0; r < 8; ++r)
            mloc[r] = fmaxf(fmaxf(sf[0][r], sf[1][r]),
                            fmaxf(sf[2][r], sf[3][r]));
        #pragma unroll
        for (int m = 8; m >= 1; m >>= 1) {
            #pragma unroll
            for (int r = 0; r < 8; ++r)
                mloc[r] = fmaxf(mloc[r], __shfl_xor(mloc[r], m, 32));
        }

        float corr[8];
        bool changed = false;
        #pragma unroll
        for (int r = 0; r < 8; ++r) {
            float mn = fmaxf(mrow[r], mloc[r]);
            if (mn > mrow[r]) changed = true;
            corr[r] = __expf(mrow[r] - mn);
            mrow[r] = mn;
            #pragma unroll
            for (int q4 = 0; q4 < 4; ++q4)
                sf[q4][r] = __expf(sf[q4][r] - mn);
        }

        if (__any(changed ? 1 : 0)) {          // FA2-style lazy rescale
            #pragma unroll
            for (int r = 0; r < 8; ++r) lrow[r] *= corr[r];
            #pragma unroll
            for (int j = 0; j < 8; ++j)
                #pragma unroll
                for (int r = 0; r < 8; ++r)
                    acc[j][r] *= corr[r];
        }

        // ---- transpose P (C-layout -> A-layout) via row-major LDS ----
        // Row-major P is exactly the A-fragment dword order: no VALU repack.
        #pragma unroll
        for (int r = 0; r < 8; ++r) {
            #pragma unroll
            for (int q4 = 0; q4 < 4; ++q4)
                pbuf[w][r + 8 * hi][16 * q4 + lo] = (__bf16)sf[q4][r];
        }
        asm volatile("s_wait_dscnt 0x0" ::: "memory");

        v16bf pa0, pa1;
        {
            v8bf p0 = *(const v8bf*)&pbuf[w][lo][16 * hi];        // t 0..31
            v8bf p1 = *(const v8bf*)&pbuf[w][lo][8 + 16 * hi];
            v8bf p2 = *(const v8bf*)&pbuf[w][lo][32 + 16 * hi];   // t 32..63
            v8bf p3 = *(const v8bf*)&pbuf[w][lo][40 + 16 * hi];
            #pragma unroll
            for (int i = 0; i < 8; ++i) {
                pa0[i] = p0[i]; pa0[8 + i] = p1[i];
                pa1[i] = p2[i]; pa1[8 + i] = p3[i];
            }
        }
        asm volatile("s_wait_dscnt 0x0" ::: "memory");

        // ---- row sums on the matrix pipe: P x ones ----
        v8f rsf = __builtin_amdgcn_wmma_f32_16x16x32_bf16(
            false, pa0, false, ones, (short)0, zero, false, false);
        rsf = __builtin_amdgcn_wmma_f32_16x16x32_bf16(
            false, pa1, false, ones, (short)0, rsf, false, false);
        #pragma unroll
        for (int r = 0; r < 8; ++r) lrow[r] += rsf[r];

        // ---- P (16x64) x V-chunk (64x16) for this wave's 8 channel tiles ----
        #pragma unroll
        for (int j = 0; j < 8; ++j) {
            const __bf16* vrow = &vtile[cur][c0 + 16 * j + lo][16 * hi];
            v16bf vf0 = *(const v16bf*)(vrow);
            v16bf vf1 = *(const v16bf*)(vrow + 32);
            acc[j] = __builtin_amdgcn_wmma_f32_16x16x32_bf16(
                false, pa0, false, vf0, (short)0, acc[j], false, false);
            acc[j] = __builtin_amdgcn_wmma_f32_16x16x32_bf16(
                false, pa1, false, vf1, (short)0, acc[j], false, false);
        }
        cur ^= 1;
    }

    // ---- epilogue: a = acc / lrow ; out = gamma*a + x ----
    const float g = gamma[0];
    const float* xb = x   + (size_t)b * SAM_C * SAM_S;
    float*       ob = out + (size_t)b * SAM_C * SAM_S;

    float rinv[8];
    #pragma unroll
    for (int r = 0; r < 8; ++r) rinv[r] = 1.0f / lrow[r];

    #pragma unroll
    for (int j = 0; j < 8; ++j) {
        const int c = c0 + 16 * j + lo;
        #pragma unroll
        for (int r = 0; r < 8; ++r) {
            const int s = s0 + r + 8 * hi;
            const size_t idx = (size_t)c * SAM_S + s;
            ob[idx] = fmaf(g, acc[j][r] * rinv[r], xb[idx]);
        }
    }
}

// ---------------------------------------------------------------------------
extern "C" void kernel_launch(void* const* d_in, const int* in_sizes, int n_in,
                              void* d_out, int out_size, void* d_ws, size_t ws_size,
                              hipStream_t stream) {
    const float* x     = (const float*)d_in[0];
    const float* w_qk  = (const float*)d_in[1];
    const float* w_v   = (const float*)d_in[2];
    const float* gamma = (const float*)d_in[3];
    float* out = (float*)d_out;

    char* ws = (char*)d_ws;
    // bf16 workspace: qs 2 MB | kt 2 MB | vc 16 MB  (total 20 MB)
    unsigned short* qs = (unsigned short*)(ws);
    unsigned short* kt = (unsigned short*)(ws + (size_t)2 * 1024 * 1024);
    unsigned short* vc = (unsigned short*)(ws + (size_t)4 * 1024 * 1024);

    dim3 g1(SAM_S / 64, 8);
    SAM_proj_kernel<<<g1, 256, 0, stream>>>(x, w_qk, w_v, qs, kt, vc);

    dim3 g2(SAM_S / 64, 8);
    SAM_attn_kernel<<<g2, 256, 0, stream>>>(x, gamma, qs, kt, vc, out);
}
